// SimpleCompressedAttention_58531814310126
// MI455X (gfx1250) — matrix operand order repaired and checked by reading, not compiled, not run on hardware
//
#include <hip/hip_runtime.h>
#include <hip/hip_bf16.h>
#include <stdint.h>

// ---------------------------------------------------------------------------
// MLA-style fused compressed attention for MI455X (gfx1250).
//   q_abs[b,h,r] = sum_d (hsq@Wq+bq)[b,h,d] * Wk[r,h*128+d]          (kernel 1)
//   p[b,h,s]   = exp(q_abs[b,h,:]·ckv[b,s,:])   (un-normalized)      (kernel 2)
//   ctx[b,h,r] = sum_s p * ckv[b,s,r] ; l[b,h] = sum_s p             (kernel 2)
//   out[b,h,d] = (ctx/l) · Wv[:,h*128+d]                             (kernel 3)
// Kernel 2 streams ckv exactly once (134 MB -> ~6us at 23.3 TB/s) with
// double-buffered TDM tiles into LDS; all matrix math is
// v_wmma_f32_16x16x32_bf16. Seq is split into 32 chunks (flash-decode style);
// partials combine linearly (no max-subtraction needed: |score| ~ O(5)).
// ---------------------------------------------------------------------------

#define NUM_HEADS   32
#define HEAD_DIM    128
#define LORA_RANK   512
#define Q_LORA_RANK 1536
#define BSZ         16
#define SEQ_LEN     4096
#define MODEL_DIM   (NUM_HEADS * HEAD_DIM)   // 4096

#define NCHUNK 32
#define CHUNK  (SEQ_LEN / NCHUNK)            // 128 seq per workgroup
#define STILE  32                            // seq tile (= WMMA K for ctx GEMM)
#define NTILE  (CHUNK / STILE)               // 4

typedef __bf16 bf16_t;
typedef __bf16        v16bf __attribute__((ext_vector_type(16)));
typedef __bf16        v4bf  __attribute__((ext_vector_type(4)));
typedef float         v8f   __attribute__((ext_vector_type(8)));
typedef float         v4f   __attribute__((ext_vector_type(4)));
typedef unsigned int  v4u   __attribute__((ext_vector_type(4)));
typedef int           v4i   __attribute__((ext_vector_type(4)));
typedef int           v8i   __attribute__((ext_vector_type(8)));

union Frag16 { v16bf v; v4u q[2]; };

// ---- LDS layout for the flash kernel (dynamic shared) ----------------------
// stag[2]: f32 [32][512]   TDM destination, double buffered        2 x 65536
// ckvN  : bf16 [32][520]   bf16 tile, normal layout (score-B)      33280
// ckvT  : bf16 [512][40]   bf16 tile, transposed   (ctx-B)         40960
// qa    : bf16 [32][520]   q_abs bf16              (score-A)       33280
// Pm    : bf16 [2][16][40] exp(scores)             (ctx-A)         2560
#define STAG_BYTES 65536
#define CKVN_ROW 520
#define CKVN_OFF  (2 * STAG_BYTES)                       // 131072
#define CKVT_ROW 40
#define CKVT_OFF  (CKVN_OFF + 32 * CKVN_ROW * 2)         // 164352
#define QA_ROW 520
#define QA_OFF    (CKVT_OFF + LORA_RANK * CKVT_ROW * 2)  // 205312
#define P_ROW 40
#define P_OFF     (QA_OFF + NUM_HEADS * QA_ROW * 2)      // 238592
#define FLASH_LDS (P_OFF + 2 * 16 * P_ROW * 2)           // 241152

__device__ __forceinline__ v8f v8f_zero() {
  v8f z;
#pragma unroll
  for (int i = 0; i < 8; ++i) z[i] = 0.0f;
  return z;
}

// TDM: async load of a [32 x 512] f32 tile (row stride 512 f32) into LDS.
// 6-arg builtin (clang-23 toolchain); trailing groups zero (2D tensor).
__device__ __forceinline__ void tdm_load_tile(unsigned lds_byte, const float* gptr) {
  uint64_t ga = (uint64_t)(uintptr_t)gptr;
  v4u g0;
  g0[0] = 1u;                                    // count=1, user descriptor
  g0[1] = lds_byte;                              // LDS byte address
  g0[2] = (unsigned)(ga & 0xFFFFFFFFu);          // global_addr[31:0]
  g0[3] = (unsigned)((ga >> 32) & 0x1FFFFFFu) | (2u << 30);  // addr[56:32]|type=2
  v8i g1;
  g1[0] = (int)(2u << 16);      // data_size = 4B
  g1[1] = (int)(512u << 16);    // tensor_dim0 = 512 (r)
  g1[2] = (int)(32u  << 16);    // tensor_dim1 = 32  (seq rows in tile)
  g1[3] = (int)(512u << 16);    // tile_dim0 = 512
  g1[4] = 32;                   // tile_dim1 = 32, tile_dim2 = 0
  g1[5] = 512;                  // tensor_dim0_stride = 512 elements
  g1[6] = 0;
  g1[7] = 0;
  v4i gz;
  gz[0] = 0; gz[1] = 0; gz[2] = 0; gz[3] = 0;
  v8i gz8;
#pragma unroll
  for (int i = 0; i < 8; ++i) gz8[i] = 0;
  __builtin_amdgcn_tensor_load_to_lds(g0, g1, gz, gz, gz8, 0);
}

// ---------------------------------------------------------------------------
// Kernel 1: q projection + Wk absorption.  grid (16, 32) x 128 threads.
// ---------------------------------------------------------------------------
__global__ __launch_bounds__(128) void mla_qabs_kernel(
    const float* __restrict__ hsq, const float* __restrict__ Wq,
    const float* __restrict__ bq,  const float* __restrict__ Wk,
    float* __restrict__ qa_ws) {
  const int b = blockIdx.x, h = blockIdx.y, tid = threadIdx.x;
  __shared__ float qh[HEAD_DIM];
  const int col = h * HEAD_DIM + tid;
  float acc = bq[col];
  const float* x = hsq + (size_t)b * Q_LORA_RANK;
  for (int i = 0; i < Q_LORA_RANK; ++i)
    acc = fmaf(x[i], Wq[(size_t)i * MODEL_DIM + col], acc);
  qh[tid] = acc;
  __syncthreads();
  for (int r = tid; r < LORA_RANK; r += 128) {
    const float* wr = Wk + (size_t)r * MODEL_DIM + h * HEAD_DIM;
    float s = 0.0f;
#pragma unroll 8
    for (int d = 0; d < HEAD_DIM; ++d) s = fmaf(qh[d], wr[d], s);
    qa_ws[((size_t)b * NUM_HEADS + h) * LORA_RANK + r] = s;
  }
}

// ---------------------------------------------------------------------------
// Kernel 2: fused streaming attention over a 128-seq chunk.
// grid (16 batches, 32 chunks) x 128 threads (4 waves).
// Waves 0/1: score GEMM + exp for head groups 0/1 (A-frags cached in regs).
// All waves: ctx GEMM, wave w -> heads (w&1)*16.., r-half (w>>1).
// ---------------------------------------------------------------------------
__global__ __launch_bounds__(128) void mla_flash_kernel(
    const float* __restrict__ ckv, const float* __restrict__ qa_ws,
    float* __restrict__ ctx_ws, float* __restrict__ l_ws) {
  extern __shared__ char smem[];
  float*  stag0 = (float*)smem;
  float*  stag1 = (float*)(smem + STAG_BYTES);
  bf16_t* ckvN  = (bf16_t*)(smem + CKVN_OFF);
  bf16_t* ckvT  = (bf16_t*)(smem + CKVT_OFF);
  bf16_t* qa    = (bf16_t*)(smem + QA_OFF);
  bf16_t* Pm    = (bf16_t*)(smem + P_OFF);

  const int b = blockIdx.x, c = blockIdx.y, tid = threadIdx.x;
  const int w    = tid >> 5;
  const int lane = tid & 31;
  const int l15  = lane & 15;
  const int hi16 = lane >> 4;   // 0: lanes 0-15, 1: lanes 16-31
  const int hg   = w & 1;       // head group for ctx GEMM
  const int rh   = w >> 1;      // r half (0: r<256, 1: r>=256)

  // Stage q_abs (bf16) for all 32 heads.
  {
    const float* src = qa_ws + (size_t)b * NUM_HEADS * LORA_RANK;
    for (int i = tid; i < NUM_HEADS * LORA_RANK; i += 128) {
      int h = i >> 9, r = i & 511;
      qa[h * QA_ROW + r] = (bf16_t)src[i];
    }
  }

  const float* gbase = ckv + ((size_t)b * SEQ_LEN + (size_t)c * CHUNK) * LORA_RANK;
  const unsigned stag_lds0 = (unsigned)(uintptr_t)stag0;
  const unsigned stag_lds1 = (unsigned)(uintptr_t)stag1;

  if (w == 0) tdm_load_tile(stag_lds0, gbase);       // prefetch tile 0
  __syncthreads();                                   // qa visible to all waves

  // Score A-fragments (q_abs rows) are tile-invariant: cache all 16 k-steps.
  Frag16 Af[16];
  if (w < 2) {
#pragma unroll
    for (int kk = 0; kk < 16; ++kk) {
      const bf16_t* ap = &qa[(w * 16 + l15) * QA_ROW + kk * 32 + (hi16 << 3)];
      Af[kk].q[0] = *(const v4u*)(ap);       // K {0..7}  (or {8..15})
      Af[kk].q[1] = *(const v4u*)(ap + 16);  // K {16..23} (or {24..31})
    }
  }
  Frag16 ones;
#pragma unroll
  for (int i = 0; i < 16; ++i) ones.v[i] = (bf16_t)1.0f;

  v8f ctx[16];
#pragma unroll
  for (int i = 0; i < 16; ++i) ctx[i] = v8f_zero();
  v8f lacc = v8f_zero();

  for (int t = 0; t < NTILE; ++t) {
    if (w == 0) {
      if (t + 1 < NTILE) {  // overlap: issue t+1, wait only for t (in-order)
        tdm_load_tile((t & 1) ? stag_lds0 : stag_lds1,
                      gbase + (size_t)(t + 1) * STILE * LORA_RANK);
        __builtin_amdgcn_s_wait_tensorcnt(1);
      } else {
        __builtin_amdgcn_s_wait_tensorcnt(0);
      }
    }
    __syncthreads();                                 // tile t resident

    // f32 tile -> bf16, written once in both layouts (4 r per thread).
    const float* sb = (t & 1) ? stag1 : stag0;
    for (int i = tid; i < STILE * (LORA_RANK / 4); i += 128) {
      const int s = i >> 7, r4 = (i & 127) << 2;
      v4f fv = *(const v4f*)(sb + s * LORA_RANK + r4);
      v4bf bv;
#pragma unroll
      for (int k = 0; k < 4; ++k) bv[k] = (bf16_t)fv[k];
      *(v4bf*)&ckvN[s * CKVN_ROW + r4] = bv;
#pragma unroll
      for (int k = 0; k < 4; ++k) ckvT[(r4 + k) * CKVT_ROW + s] = bv[k];
    }
    __syncthreads();

    // ---- score GEMM: S[16h x 32s] = qa[16h x 512r] * ckv^T, waves 0/1 ----
    if (w < 2) {
      v8f sc0 = v8f_zero(), sc1 = v8f_zero();
#pragma unroll
      for (int kk = 0; kk < 16; ++kk) {              // K = r, 32 per step
#pragma unroll
        for (int nt = 0; nt < 2; ++nt) {             // two 16-col N tiles
          const bf16_t* bp =
              &ckvN[(nt * 16 + l15) * CKVN_ROW + kk * 32 + (hi16 << 3)];
          Frag16 B;                                  // K {0..7,16..23} (or {8..15,24..31})
          B.q[0] = *(const v4u*)(bp);
          B.q[1] = *(const v4u*)(bp + 16);
          if (nt == 0)
            sc0 = __builtin_amdgcn_wmma_f32_16x16x32_bf16(false, Af[kk].v, false, B.v,
                                                          (short)0, sc0, false, false);
          else
            sc1 = __builtin_amdgcn_wmma_f32_16x16x32_bf16(false, Af[kk].v, false, B.v,
                                                          (short)0, sc1, false, false);
        }
      }
      // exp (no max-subtraction; chunks combine linearly) -> P in LDS
#pragma unroll
      for (int j = 0; j < 8; ++j) {
        int h = j + (hi16 << 3);
        Pm[(w * 16 + h) * P_ROW + l15]      = (bf16_t)__expf(sc0[j]);
        Pm[(w * 16 + h) * P_ROW + 16 + l15] = (bf16_t)__expf(sc1[j]);
      }
    }
    __syncthreads();

    // ---- ctx GEMM: ctx[16h x 16r] += P[16h x 32s] * ckv[32s x 16r] ----
    Frag16 Pf;                                       // A-frag gap layout
    const bf16_t* pp = &Pm[(hg * 16 + l15) * P_ROW + (hi16 << 3)];
    Pf.q[0] = *(const v4u*)(pp);
    Pf.q[1] = *(const v4u*)(pp + 16);
    if (w < 2)  // row-sum of probs via ones-B (l[h] = sum_s p)
      lacc = __builtin_amdgcn_wmma_f32_16x16x32_bf16(false, Pf.v, false, ones.v,
                                                     (short)0, lacc, false, false);
#pragma unroll
    for (int i = 0; i < 16; ++i) {
      const int r = (rh * 16 + i) * 16 + l15;        // this lane's r column
      const bf16_t* bp = &ckvT[r * CKVT_ROW + (hi16 << 3)];
      Frag16 B;                                      // K {0..7,16..23} (or {8..15,24..31})
      B.q[0] = *(const v4u*)(bp);
      B.q[1] = *(const v4u*)(bp + 16);
      ctx[i] = __builtin_amdgcn_wmma_f32_16x16x32_bf16(false, Pf.v, false, B.v,
                                                       (short)0, ctx[i], false, false);
    }
    __syncthreads();                                 // ckvN/ckvT free for t+1
  }

  // Store partial ctx (C-frag layout: VGPR j -> head j+8*hi16, lane%16 -> r col).
  {
    float* cbase = ctx_ws +
        (((size_t)b * NCHUNK + c) * NUM_HEADS + hg * 16 + (hi16 << 3)) * LORA_RANK;
#pragma unroll
    for (int i = 0; i < 16; ++i) {
      const int rcol = (rh * 16 + i) * 16 + l15;
#pragma unroll
      for (int j = 0; j < 8; ++j)
        cbase[(size_t)j * LORA_RANK + rcol] = ctx[i][j];
    }
    if (w < 2 && l15 == 0) {
      float* lbase = l_ws + ((size_t)b * NCHUNK + c) * NUM_HEADS + w * 16 + (hi16 << 3);
#pragma unroll
      for (int j = 0; j < 8; ++j) lbase[j] = lacc[j];
    }
  }
}

// ---------------------------------------------------------------------------
// Kernel 3: combine chunk partials + Wv projection.  grid (16,32) x 128.
// ---------------------------------------------------------------------------
__global__ __launch_bounds__(128) void mla_out_kernel(
    const float* __restrict__ ctx_ws, const float* __restrict__ l_ws,
    const float* __restrict__ Wv, float* __restrict__ out) {
  const int b = blockIdx.x, h = blockIdx.y, tid = threadIdx.x;
  __shared__ float cn[LORA_RANK];
  float L = 0.0f;
  for (int c = 0; c < NCHUNK; ++c)
    L += l_ws[((size_t)b * NCHUNK + c) * NUM_HEADS + h];
  const float invL = 1.0f / L;
  for (int r = tid; r < LORA_RANK; r += 128) {
    float s = 0.0f;
    for (int c = 0; c < NCHUNK; ++c)
      s += ctx_ws[(((size_t)b * NCHUNK + c) * NUM_HEADS + h) * LORA_RANK + r];
    cn[r] = s * invL;
  }
  __syncthreads();
  const int col = h * HEAD_DIM + tid;
  float acc = 0.0f;
#pragma unroll 8
  for (int r = 0; r < LORA_RANK; ++r)
    acc = fmaf(cn[r], Wv[(size_t)r * MODEL_DIM + col], acc);
  out[(size_t)b * MODEL_DIM + col] = acc;
}

// ---------------------------------------------------------------------------
extern "C" void kernel_launch(void* const* d_in, const int* in_sizes, int n_in,
                              void* d_out, int out_size, void* d_ws, size_t ws_size,
                              hipStream_t stream) {
  const float* hsq = (const float*)d_in[0];
  const float* ckv = (const float*)d_in[1];
  const float* Wq  = (const float*)d_in[2];
  const float* bq  = (const float*)d_in[3];
  const float* Wk  = (const float*)d_in[4];
  const float* Wv  = (const float*)d_in[5];
  float* out = (float*)d_out;

  char* ws = (char*)d_ws;
  const size_t qa_bytes  = (size_t)BSZ * NUM_HEADS * LORA_RANK * sizeof(float);
  const size_t ctx_bytes = (size_t)BSZ * NCHUNK * NUM_HEADS * LORA_RANK * sizeof(float);
  float* qa_ws  = (float*)ws;
  float* ctx_ws = (float*)(ws + qa_bytes);
  float* l_ws   = (float*)(ws + qa_bytes + ctx_bytes);

  mla_qabs_kernel<<<dim3(BSZ, NUM_HEADS), 128, 0, stream>>>(hsq, Wq, bq, Wk, qa_ws);
  mla_flash_kernel<<<dim3(BSZ, NCHUNK), 128, FLASH_LDS, stream>>>(ckv, qa_ws,
                                                                  ctx_ws, l_ws);
  mla_out_kernel<<<dim3(BSZ, NUM_HEADS), 128, 0, stream>>>(ctx_ws, l_ws, Wv, out);
}
